// QuantumGeometricAttention_28484223108084
// MI455X (gfx1250) — matrix operand order repaired and checked by reading, not compiled, not run on hardware
//
#include <hip/hip_runtime.h>
#include <hip/hip_bf16.h>

#define B_  8
#define S_  1024
#define Dm  1024
#define H_  8
#define HD  128
#define M_  64
#define HM  512      /* H_*M_ */
#define EPSF 1e-7f
#define DT_  0.1f
#define FLOW_STEPS_ 10

typedef __attribute__((ext_vector_type(16))) _Float16 v16h;
typedef __attribute__((ext_vector_type(8)))  _Float16 v8h;
typedef __attribute__((ext_vector_type(8)))  float    v8f;
typedef unsigned int u32x4 __attribute__((ext_vector_type(4)));
typedef int          i32x4 __attribute__((ext_vector_type(4)));
typedef int          i32x8 __attribute__((ext_vector_type(8)));

#define DEVFN static __device__ __forceinline__

DEVFN v8f wmma_f16(v16h a, v16h b, v8f c) {
  return __builtin_amdgcn_wmma_f32_16x16x32_f16(false, a, false, b, (short)0, c,
                                                false, false);
}

DEVFN v8f zero8() {
  v8f z;
#pragma unroll
  for (int i = 0; i < 8; ++i) z[i] = 0.0f;
  return z;
}

// A fragment (16x32 f16): two contiguous 16B runs -> 2x ds_load_b128.
DEVFN v16h load_a(const _Float16* p, int ld, int kbase, int lane) {
  const int Mr   = lane & 15;
  const int koff = (lane >> 4) << 3;
  const _Float16* q = p + Mr * ld + kbase + koff;
  v8h lo = *(const v8h*)q;
  v8h hi = *(const v8h*)(q + 16);
  v16h a;
#pragma unroll
  for (int i = 0; i < 8; ++i) { a[i] = lo[i]; a[i + 8] = hi[i]; }
  return a;
}

// B fragment (32x16 f16) from TRANSPOSED source (B[k][n] = p[n*ld + k]):
// two contiguous 16B runs (k = 8h..8h+7, 16+8h..16+8h+7) -> 2x ds_load_b128.
DEVFN v16h load_bt(const _Float16* p, int ld, int kbase, int nbase, int lane) {
  const int Nc   = lane & 15;
  const int koff = (lane >> 4) << 3;
  const _Float16* q = p + (nbase + Nc) * ld + kbase + koff;
  v8h lo = *(const v8h*)q;
  v8h hi = *(const v8h*)(q + 16);
  v16h b;
#pragma unroll
  for (int i = 0; i < 8; ++i) { b[i] = lo[i]; b[i + 8] = hi[i]; }
  return b;
}

// Complex GEMM, 3-accumulator form (no per-kblock negation):
//   accP += Ar*Br ; accM += Ai*Bi ; accI += Ar*Bi + Ai*Br
// Final: real = accP - accM ; imag = accI
DEVFN void cgemm3(v8f& accP, v8f& accM, v8f& accI,
                  const _Float16* Ar, const _Float16* Ai, int lda, int kbaseA,
                  const _Float16* BrT, const _Float16* BiT, int ldb, int kbaseB,
                  int nbase, int lane) {
  v16h ar = load_a(Ar, lda, kbaseA, lane);
  v16h ai = load_a(Ai, lda, kbaseA, lane);
  v16h br = load_bt(BrT, ldb, kbaseB, nbase, lane);
  v16h bi = load_bt(BiT, ldb, kbaseB, nbase, lane);
  accP = wmma_f16(ar, br, accP);
  accM = wmma_f16(ai, bi, accM);
  accI = wmma_f16(ar, bi, accI);
  accI = wmma_f16(ai, br, accI);
}

// ---------------------------------------------------------------------------
// Tensor Data Mover: 2D tile (f16 elements) global -> LDS.
// ---------------------------------------------------------------------------
#if __has_builtin(__builtin_amdgcn_tensor_load_to_lds)
#define USE_TDM 1
#warning "QGA: tensor_load_to_lds ENABLED"
#else
#define USE_TDM 0
#warning "QGA: tensor_load_to_lds NOT available - manual staging fallback"
#endif

#if USE_TDM
DEVFN void tdm_load_2d(unsigned lds_off, const _Float16* gptr,
                       int tile_w, int tile_h, int row_stride) {
  unsigned long long ga = (unsigned long long)(size_t)gptr;
  u32x4 g0;
  g0[0] = 1u;                                   // count=1, user descriptor
  g0[1] = lds_off;                              // lds_addr [63:32]
  g0[2] = (unsigned)(ga & 0xffffffffull);       // global_addr [95:64]
  g0[3] = (unsigned)((ga >> 32) & 0x01ffffffull) | (2u << 30);  // addr hi | type=2
  i32x8 g1;
  g1[0] = 1 << 16;                              // data_size = 2 bytes
  g1[1] = (tile_w & 0xffff) << 16;              // tensor_dim0 lo16 @ bits 63:48
  g1[2] = ((unsigned)tile_w >> 16) | ((tile_h & 0xffff) << 16);
  g1[3] = ((unsigned)tile_h >> 16) | ((tile_w & 0xffff) << 16);  // | tile_dim0
  g1[4] = tile_h & 0xffff;                      // tile_dim1 (tile_dim2 = 0)
  g1[5] = row_stride;                           // tensor_dim0_stride [31:0]
  g1[6] = 0;
  g1[7] = 0;
  i32x4 gz; gz[0] = 0; gz[1] = 0; gz[2] = 0; gz[3] = 0;
#if __clang_major__ >= 23
  i32x8 gz8;
#pragma unroll
  for (int i = 0; i < 8; ++i) gz8[i] = 0;
  __builtin_amdgcn_tensor_load_to_lds(g0, g1, gz, gz, gz8, 0);
#else
  __builtin_amdgcn_tensor_load_to_lds(g0, g1, gz, gz, 0);
#endif
}
#endif

// ---------------------------------------------------------------------------
// Kernel 1: manifold projection + exp/log map + q/k/v (one wave / 16 s-rows)
// ---------------------------------------------------------------------------
__global__ __launch_bounds__(32)
void qga_proj_qkv(const float* __restrict__ xr, const float* __restrict__ xi,
                  const float* __restrict__ Wp, const float* __restrict__ bp,
                  const float* __restrict__ Wq, const float* __restrict__ bq,
                  const float* __restrict__ Wk, const float* __restrict__ bk,
                  const float* __restrict__ Wv, const float* __restrict__ bv,
                  const float* __restrict__ metric,
                  _Float16* __restrict__ qor, _Float16* __restrict__ qoi,
                  _Float16* __restrict__ kor, _Float16* __restrict__ koi,
                  _Float16* __restrict__ vor_, _Float16* __restrict__ voi) {
  __shared__ _Float16 WpR[M_ * HD], WpI[M_ * HD];  // [m][d] straight copy
  __shared__ _Float16 WbR[M_ * M_], WbI[M_ * M_];  // reusable [n][m]
  __shared__ _Float16 XR[16 * HD], XI[16 * HD];
  __shared__ _Float16 UR[16 * M_], UI[16 * M_];

  const int lane  = threadIdx.x & 31;
  const int bid   = blockIdx.x;
  const int b     = bid / (H_ * (S_ / 16));
  const int r0    = bid % (H_ * (S_ / 16));
  const int h     = r0 / (S_ / 16);
  const int sbase = (r0 % (S_ / 16)) * 16;

  for (int idx = lane; idx < M_ * HD; idx += 32) {
    WpR[idx] = (_Float16)Wp[idx * 2 + 0];
    WpI[idx] = (_Float16)Wp[idx * 2 + 1];
  }
  for (int idx = lane; idx < 16 * HD; idx += 32) {
    int r = idx >> 7, d = idx & 127;
    size_t g = ((size_t)(b * S_ + sbase + r)) * Dm + h * HD + d;
    XR[idx] = (_Float16)xr[g];
    XI[idx] = (_Float16)xi[g];
  }
  __syncthreads();

  const int hi = lane >> 4, col0 = lane & 15;
  float vre[4][8], vim[4][8], sc[8];

  // v0 = x * Wp^T + bp
#pragma unroll
  for (int nt = 0; nt < 4; ++nt) {
    v8f aP = zero8(), aM = zero8(), aI = zero8();
#pragma unroll
    for (int kb = 0; kb < 4; ++kb)
      cgemm3(aP, aM, aI, XR, XI, HD, kb * 32, WpR, WpI, HD, kb * 32, nt * 16, lane);
    v8f rr = aP - aM;
    int col = nt * 16 + col0;
    float br = bp[col * 2 + 0], bi = bp[col * 2 + 1];
#pragma unroll
    for (int v = 0; v < 8; ++v) { vre[nt][v] = rr[v] + br; vim[nt][v] = aI[v] + bi; }
  }

  // exp map
#pragma unroll
  for (int v = 0; v < 8; ++v) {
    float s = 0.f;
#pragma unroll
    for (int nt = 0; nt < 4; ++nt) s += vre[nt][v] * vre[nt][v] + vim[nt][v] * vim[nt][v];
#pragma unroll
    for (int m = 1; m < 16; m <<= 1) s += __shfl_xor(s, m, 32);
    float n = sqrtf(s + EPSF);
    sc[v] = tanhf(n) / n;
  }
#pragma unroll
  for (int nt = 0; nt < 4; ++nt)
#pragma unroll
    for (int v = 0; v < 8; ++v) {
      int row = v + (hi << 3), col = nt * 16 + col0;
      UR[row * M_ + col] = (_Float16)(vre[nt][v] * sc[v]);
      UI[row * M_ + col] = (_Float16)(vim[nt][v] * sc[v]);
    }

  for (int idx = lane; idx < M_ * M_; idx += 32) {
    WbR[idx] = (_Float16)metric[idx * 2 + 0];
    WbI[idx] = (_Float16)metric[idx * 2 + 1];
  }
  __syncthreads();

  // t = h * metric^T
#pragma unroll
  for (int nt = 0; nt < 4; ++nt) {
    v8f aP = zero8(), aM = zero8(), aI = zero8();
#pragma unroll
    for (int kb = 0; kb < 2; ++kb)
      cgemm3(aP, aM, aI, UR, UI, M_, kb * 32, WbR, WbI, M_, kb * 32, nt * 16, lane);
    v8f rr = aP - aM;
#pragma unroll
    for (int v = 0; v < 8; ++v) { vre[nt][v] = rr[v]; vim[nt][v] = aI[v]; }
  }
  // log map
#pragma unroll
  for (int v = 0; v < 8; ++v) {
    float s = 0.f;
#pragma unroll
    for (int nt = 0; nt < 4; ++nt) s += vre[nt][v] * vre[nt][v] + vim[nt][v] * vim[nt][v];
#pragma unroll
    for (int m = 1; m < 16; m <<= 1) s += __shfl_xor(s, m, 32);
    float n = sqrtf(s + EPSF);
    float c = fminf(n, 1.0f - 1e-5f);
    sc[v] = atanhf(c) / n;
  }
  __syncthreads();
#pragma unroll
  for (int nt = 0; nt < 4; ++nt)
#pragma unroll
    for (int v = 0; v < 8; ++v) {
      int row = v + (hi << 3), col = nt * 16 + col0;
      UR[row * M_ + col] = (_Float16)(vre[nt][v] * sc[v]);
      UI[row * M_ + col] = (_Float16)(vim[nt][v] * sc[v]);
    }

  // q / k / v projections (v stored transposed: vT[b,h,m,s])
  auto proj_one = [&](const float* W, const float* bb, _Float16* gR,
                      _Float16* gI, bool vtrans) {
    __syncthreads();
    for (int idx = lane; idx < M_ * M_; idx += 32) {
      WbR[idx] = (_Float16)W[idx * 2 + 0];
      WbI[idx] = (_Float16)W[idx * 2 + 1];
    }
    __syncthreads();
#pragma unroll
    for (int nt = 0; nt < 4; ++nt) {
      v8f aP = zero8(), aM = zero8(), aI = zero8();
#pragma unroll
      for (int kb = 0; kb < 2; ++kb)
        cgemm3(aP, aM, aI, UR, UI, M_, kb * 32, WbR, WbI, M_, kb * 32, nt * 16, lane);
      v8f rr = aP - aM;
      int col = nt * 16 + col0;
      float brv = bb[col * 2 + 0], biv = bb[col * 2 + 1];
#pragma unroll
      for (int v = 0; v < 8; ++v) {
        int s = sbase + v + (hi << 3);
        size_t g;
        if (vtrans)
          g = ((size_t)(b * H_ + h) * M_ + col) * S_ + s;
        else
          g = (((size_t)(b * S_ + s)) * H_ + h) * M_ + col;
        gR[g] = (_Float16)(rr[v] + brv);
        gI[g] = (_Float16)(aI[v] + biv);
      }
    }
  };
  proj_one(Wq, bq, qor, qoi, false);
  proj_one(Wk, bk, kor, koi, false);
  proj_one(Wv, bv, vor_, voi, true);
}

// ---------------------------------------------------------------------------
// Kernel 2: flash attention; double-buffered TDM staging of K/V tiles
// ---------------------------------------------------------------------------
__global__ __launch_bounds__(128)
void qga_attn(const _Float16* __restrict__ qr_g, const _Float16* __restrict__ qi_g,
              const _Float16* __restrict__ kr_g, const _Float16* __restrict__ ki_g,
              const _Float16* __restrict__ vtr_g, const _Float16* __restrict__ vti_g,
              float* __restrict__ zout) {
  __shared__ _Float16 KR[2][32 * M_], KI[2][32 * M_];    // [krow][m]
  __shared__ _Float16 VTR[2][M_ * 32], VTI[2][M_ * 32];  // [m][krow]
  __shared__ _Float16 Pb[4][16 * 32];

  const int tid = threadIdx.x, lane = tid & 31, w = tid >> 5;
  const int bid = blockIdx.x;
  const int b   = bid / (H_ * (S_ / 64));
  const int r0  = bid % (H_ * (S_ / 64));
  const int h   = r0 / (S_ / 64);
  const int sq  = (r0 % (S_ / 64)) * 64 + w * 16;
  const size_t base  = (size_t)(b * S_) * HM + h * M_;
  const size_t vbase = (size_t)(b * H_ + h) * M_ * S_;

  // q fragments: A = [qr | qi], K = 128
  v16h qf[4];
  {
    const int Mr = lane & 15;
    const _Float16* qrp = qr_g + base + (size_t)(sq + Mr) * HM;
    const _Float16* qip = qi_g + base + (size_t)(sq + Mr) * HM;
#pragma unroll
    for (int kb = 0; kb < 4; ++kb) {
      const _Float16* p = (kb < 2) ? qrp : qip;
      qf[kb] = load_a(p, 0, (kb & 1) * 32, lane);
    }
  }

  float mrun[8], lrun[8];
  v8f ctr[4], cti[4];
#pragma unroll
  for (int v = 0; v < 8; ++v) { mrun[v] = -1e30f; lrun[v] = 0.f; }
#pragma unroll
  for (int nt = 0; nt < 4; ++nt) { ctr[nt] = zero8(); cti[nt] = zero8(); }

  const float scl = 0.125f;
  const int hi = lane >> 4, col0 = lane & 15;

  auto stage = [&](int kb32, int buf) {
    const size_t kgo = base + (size_t)(kb32 * 32) * HM;
    const size_t vgo = vbase + (size_t)(kb32 * 32);
#if USE_TDM
    if (w == 0)
      tdm_load_2d((unsigned)(size_t)(void*)&KR[buf][0], kr_g + kgo, 64, 32, HM);
    else if (w == 1)
      tdm_load_2d((unsigned)(size_t)(void*)&KI[buf][0], ki_g + kgo, 64, 32, HM);
    else if (w == 2)
      tdm_load_2d((unsigned)(size_t)(void*)&VTR[buf][0], vtr_g + vgo, 32, 64, S_);
    else
      tdm_load_2d((unsigned)(size_t)(void*)&VTI[buf][0], vti_g + vgo, 32, 64, S_);
#else
    for (int idx = tid; idx < 32 * M_; idx += 128) {
      KR[buf][idx]  = kr_g[kgo + (size_t)(idx >> 6) * HM + (idx & 63)];
      KI[buf][idx]  = ki_g[kgo + (size_t)(idx >> 6) * HM + (idx & 63)];
      VTR[buf][idx] = vtr_g[vgo + (size_t)(idx >> 5) * S_ + (idx & 31)];
      VTI[buf][idx] = vti_g[vgo + (size_t)(idx >> 5) * S_ + (idx & 31)];
    }
#endif
  };

  stage(0, 0);
  for (int kb32 = 0; kb32 < S_ / 32; ++kb32) {
    const int cur = kb32 & 1;
#if USE_TDM
    __builtin_amdgcn_s_wait_tensorcnt((short)0);
#endif
    __syncthreads();
    if (kb32 + 1 < S_ / 32) stage(kb32 + 1, cur ^ 1);

    // scores: B[k=feature][n=krow] = Ktile[krow][feature]
    v8f s0 = zero8(), s1 = zero8();
#pragma unroll
    for (int kb = 0; kb < 4; ++kb) {
      const _Float16* src = (kb < 2) ? &KR[cur][0] : &KI[cur][0];
      int kbase = (kb & 1) * 32;
      v16h b0 = load_bt(src, M_, kbase, 0, lane);
      v16h b1 = load_bt(src, M_, kbase, 16, lane);
      s0 = wmma_f16(qf[kb], b0, s0);
      s1 = wmma_f16(qf[kb], b1, s1);
    }

    // online softmax
    float al[8];
#pragma unroll
    for (int v = 0; v < 8; ++v) {
      float a0 = s0[v] * scl, a1 = s1[v] * scl;
      float mx = fmaxf(a0, a1);
#pragma unroll
      for (int m = 1; m < 16; m <<= 1) mx = fmaxf(mx, __shfl_xor(mx, m, 32));
      float mn = fmaxf(mrun[v], mx);
      float alpha = __expf(mrun[v] - mn);
      float p0 = __expf(a0 - mn), p1 = __expf(a1 - mn);
      int row = v + (hi << 3);
      Pb[w][row * 32 + col0]      = (_Float16)p0;
      Pb[w][row * 32 + 16 + col0] = (_Float16)p1;
      float ps = p0 + p1;
#pragma unroll
      for (int m = 1; m < 16; m <<= 1) ps += __shfl_xor(ps, m, 32);
      lrun[v] = lrun[v] * alpha + ps;
      mrun[v] = mn;
      al[v] = alpha;
    }
#pragma unroll
    for (int nt = 0; nt < 4; ++nt)
#pragma unroll
      for (int v = 0; v < 8; ++v) { ctr[nt][v] *= al[v]; cti[nt][v] *= al[v]; }

    // ctx += P x V : B[k=krow][n=m] = VT[m][krow]
    v16h pa = load_a(&Pb[w][0], 32, 0, lane);
#pragma unroll
    for (int nt = 0; nt < 4; ++nt) {
      v16h bvr = load_bt(&VTR[cur][0], 32, 0, nt * 16, lane);
      v16h bvi = load_bt(&VTI[cur][0], 32, 0, nt * 16, lane);
      ctr[nt] = wmma_f16(pa, bvr, ctr[nt]);
      cti[nt] = wmma_f16(pa, bvi, cti[nt]);
    }
  }

#pragma unroll
  for (int nt = 0; nt < 4; ++nt)
#pragma unroll
    for (int v = 0; v < 8; ++v) {
      int s = sq + v + (hi << 3);
      float inv = (lrun[v] > 0.f) ? 1.f / lrun[v] : 0.f;
      size_t g = base + (size_t)s * HM + nt * 16 + col0;
      zout[2 * g + 0] = ctr[nt][v] * inv;
      zout[2 * g + 1] = cti[nt][v] * inv;
    }
}

// ---------------------------------------------------------------------------
// Kernel 3: 10-step Euler geometric flow  z += dt * ctanh(z * metric^T)
// ---------------------------------------------------------------------------
__global__ __launch_bounds__(128)
void qga_flow(const float* __restrict__ metric, float* __restrict__ z) {
  __shared__ _Float16 MR[M_ * M_], MI[M_ * M_];
  __shared__ _Float16 ZR[4][16 * M_], ZI[4][16 * M_];
  const int tid = threadIdx.x, lane = tid & 31, w = tid >> 5;
  const int rbase = blockIdx.x * 64 + w * 16;

  for (int idx = tid; idx < M_ * M_; idx += 128) {
    MR[idx] = (_Float16)metric[idx * 2 + 0];
    MI[idx] = (_Float16)metric[idx * 2 + 1];
  }
  __syncthreads();

  const int hi = lane >> 4, col0 = lane & 15;
  float zr[4][8], zi[4][8];
#pragma unroll
  for (int nt = 0; nt < 4; ++nt)
#pragma unroll
    for (int v = 0; v < 8; ++v) {
      size_t g = (size_t)(rbase + v + (hi << 3)) * M_ + nt * 16 + col0;
      zr[nt][v] = z[2 * g + 0];
      zi[nt][v] = z[2 * g + 1];
    }

  for (int step = 0; step < FLOW_STEPS_; ++step) {
#pragma unroll
    for (int nt = 0; nt < 4; ++nt)
#pragma unroll
      for (int v = 0; v < 8; ++v) {
        int row = v + (hi << 3), col = nt * 16 + col0;
        ZR[w][row * M_ + col] = (_Float16)zr[nt][v];
        ZI[w][row * M_ + col] = (_Float16)zi[nt][v];
      }
#pragma unroll
    for (int nt = 0; nt < 4; ++nt) {
      v8f aP = zero8(), aM = zero8(), aI = zero8();
#pragma unroll
      for (int kb = 0; kb < 2; ++kb)
        cgemm3(aP, aM, aI, &ZR[w][0], &ZI[w][0], M_, kb * 32, MR, MI, M_, kb * 32,
               nt * 16, lane);
      v8f rr = aP - aM;
#pragma unroll
      for (int v = 0; v < 8; ++v) {
        float ta = tanhf(rr[v]);
        float tb = tanf(aI[v]);
        float t2 = tb * tb;
        float den = 1.0f + ta * ta * t2;
        zr[nt][v] += DT_ * (ta * (1.0f + t2) / den);
        zi[nt][v] += DT_ * (tb * (1.0f - ta * ta) / den);
      }
    }
  }

#pragma unroll
  for (int nt = 0; nt < 4; ++nt)
#pragma unroll
    for (int v = 0; v < 8; ++v) {
      size_t g = (size_t)(rbase + v + (hi << 3)) * M_ + nt * 16 + col0;
      z[2 * g + 0] = zr[nt][v];
      z[2 * g + 1] = zi[nt][v];
    }
}

// ---------------------------------------------------------------------------
// Kernel 4: out = z * Wo^T + bo + (z_h * Wpi^T + bpi)
// ---------------------------------------------------------------------------
__global__ __launch_bounds__(64)
void qga_out(const float* __restrict__ zv, const float* __restrict__ Wo,
             const float* __restrict__ bo, const float* __restrict__ Wpi,
             const float* __restrict__ bpi, float* __restrict__ out) {
  __shared__ _Float16 ZR[16 * HM], ZI[16 * HM];      // 16 rows x 512
  __shared__ _Float16 WSR[128 * 32], WSI[128 * 32];  // [n=d][k] transposed tile
  const int tid = threadIdx.x, lane = tid & 31, w = tid >> 5;
  const int rt = blockIdx.x >> 3;
  const int cb = blockIdx.x & 7;   // 128-col block == head index
  const int rowbase = rt * 16;

  for (int idx = tid; idx < 16 * HM; idx += 64) {
    int r = idx >> 9, f = idx & 511;
    size_t g = ((size_t)(rowbase + r)) * HM + f;
    ZR[idx] = (_Float16)zv[2 * g + 0];
    ZI[idx] = (_Float16)zv[2 * g + 1];
  }

  v8f aP[4], aM[4], aI[4];
#pragma unroll
  for (int nt = 0; nt < 4; ++nt) { aP[nt] = zero8(); aM[nt] = zero8(); aI[nt] = zero8(); }

  // main projection over K = 512
  for (int kb = 0; kb < 16; ++kb) {
    __syncthreads();
    for (int idx = tid; idx < 128 * 32; idx += 64) {
      int dl = idx >> 5, kk = idx & 31;
      size_t g = ((size_t)(cb * 128 + dl) * HM + kb * 32 + kk) * 2;
      WSR[idx] = (_Float16)Wo[g + 0];
      WSI[idx] = (_Float16)Wo[g + 1];
      if (kb + 1 < 16) __builtin_prefetch(Wo + g + 64 * 2, 0, 1);
    }
    __syncthreads();
#pragma unroll
    for (int nt = 0; nt < 4; ++nt)
      cgemm3(aP[nt], aM[nt], aI[nt], ZR, ZI, HM, kb * 32, WSR, WSI, 32, 0,
             w * 64 + nt * 16, lane);
  }

  // back-projection (per-head slice of z, K = 64)
  for (int kb = 0; kb < 2; ++kb) {
    __syncthreads();
    for (int idx = tid; idx < 128 * 32; idx += 64) {
      int dl = idx >> 5, kk = idx & 31;
      size_t g = ((size_t)dl * M_ + kb * 32 + kk) * 2;
      WSR[idx] = (_Float16)Wpi[g + 0];
      WSI[idx] = (_Float16)Wpi[g + 1];
    }
    __syncthreads();
#pragma unroll
    for (int nt = 0; nt < 4; ++nt)
      cgemm3(aP[nt], aM[nt], aI[nt], ZR, ZI, HM, cb * M_ + kb * 32, WSR, WSI, 32, 0,
             w * 64 + nt * 16, lane);
  }

  const int hi = lane >> 4, col0 = lane & 15;
#pragma unroll
  for (int nt = 0; nt < 4; ++nt) {
    v8f rr = aP[nt] - aM[nt];
#pragma unroll
    for (int v = 0; v < 8; ++v) {
      int row = rowbase + v + (hi << 3);
      int dl = w * 64 + nt * 16 + col0;
      int d = cb * 128 + dl;
      size_t g = (size_t)row * Dm + d;
      out[2 * g + 0] = rr[v] + bo[2 * d + 0] + bpi[2 * dl + 0];
      out[2 * g + 1] = aI[nt][v] + bo[2 * d + 1] + bpi[2 * dl + 1];
    }
  }
}

// ---------------------------------------------------------------------------
extern "C" void kernel_launch(void* const* d_in, const int* in_sizes, int n_in,
                              void* d_out, int out_size, void* d_ws,
                              size_t ws_size, hipStream_t stream) {
  (void)in_sizes; (void)n_in; (void)out_size; (void)ws_size;
  const float* xr  = (const float*)d_in[0];
  const float* xi  = (const float*)d_in[1];
  const float* Wp  = (const float*)d_in[2];
  const float* bp  = (const float*)d_in[3];
  const float* Wq  = (const float*)d_in[4];
  const float* bq  = (const float*)d_in[5];
  const float* Wk  = (const float*)d_in[6];
  const float* bk  = (const float*)d_in[7];
  const float* Wv  = (const float*)d_in[8];
  const float* bv  = (const float*)d_in[9];
  const float* Wpi = (const float*)d_in[10];
  const float* bpi = (const float*)d_in[11];
  const float* Wo  = (const float*)d_in[12];
  const float* bo  = (const float*)d_in[13];
  const float* metric = (const float*)d_in[14];

  const size_t NQ = (size_t)B_ * S_ * H_ * M_;
  _Float16* qr = (_Float16*)d_ws;
  _Float16* qi = qr + NQ;
  _Float16* kr = qi + NQ;
  _Float16* ki = kr + NQ;
  _Float16* vt_r = ki + NQ;   // transposed: [b,h,m,s]
  _Float16* vt_i = vt_r + NQ;
  float* z = (float*)(vt_i + NQ);  // float2 per element, NQ elements

  qga_proj_qkv<<<dim3(B_ * H_ * (S_ / 16)), dim3(32), 0, stream>>>(
      xr, xi, Wp, bp, Wq, bq, Wk, bk, Wv, bv, metric, qr, qi, kr, ki, vt_r, vt_i);
  qga_attn<<<dim3(B_ * H_ * (S_ / 64)), dim3(128), 0, stream>>>(
      qr, qi, kr, ki, vt_r, vt_i, z);
  qga_flow<<<dim3((B_ * S_ * H_) / 64), dim3(128), 0, stream>>>(metric, z);
  qga_out<<<dim3((B_ * S_ / 16) * 8), dim3(64), 0, stream>>>(
      z, Wo, bo, Wpi, bpi, (float*)d_out);
}
